// S4DBlock_59863254171678
// MI455X (gfx1250) — matrix operand order repaired and checked by reading, not compiled, not run on hardware
//
#include <hip/hip_runtime.h>
#include <hip/hip_bf16.h>
#include <math.h>

#define BB 8
#define CC 256
#define TT 16000
#define NH 32
#define NCHUNK 16
#define LCHUNK (TT / NCHUNK)   /* 1000 */
#define NTOT (BB * TT)         /* 128000 */

typedef __attribute__((ext_vector_type(16))) __bf16 v16bf;
typedef __attribute__((ext_vector_type(8)))  float  v8f;

__device__ __forceinline__ unsigned short f2bf(float f) {
    unsigned u = __float_as_uint(f);
    u = u + 0x7FFFu + ((u >> 16) & 1u);
    return (unsigned short)(u >> 16);
}
__device__ __forceinline__ float bf2f(unsigned short h) {
    return __uint_as_float(((unsigned)h) << 16);
}

// ---------------- S4D mode parameters (per c,n): w, 2*Cc, w^LCHUNK -------------
__global__ void k_params(const float* log_dt, const float* log_A_real,
                         const float* A_imag, const float* C_re, const float* C_im,
                         float4* pw, float2* pwL) {
    int id = blockIdx.x * blockDim.x + threadIdx.x;   // c*32 + n
    if (id >= CC * NH) return;
    int c = id >> 5;
    float dt = expf(log_dt[c]);
    float Ar = -expf(log_A_real[id]);
    float Ai = A_imag[id];
    float dr = dt * Ar, di = dt * Ai;
    float e  = expf(dr);
    float wr = e * cosf(di);
    float wi = e * sinf(di);
    float nr = wr - 1.0f, ni = wi;
    float den = 1.0f / (Ar * Ar + Ai * Ai);
    float qr = (nr * Ar + ni * Ai) * den;
    float qi = (ni * Ar - nr * Ai) * den;
    float cr = C_re[id], ci = C_im[id];
    float ccr = 2.0f * (cr * qr - ci * qi);
    float cci = 2.0f * (cr * qi + ci * qr);
    pw[id] = make_float4(wr, wi, ccr, cci);
    float eL  = expf(dr * (float)LCHUNK);
    float ang = di * (float)LCHUNK;
    pwL[id] = make_float2(eL * cosf(ang), eL * sinf(ang));
}

// ---------------- W_out f32 -> bf16 -------------------------------------------
__global__ void k_wcvt(const float* W, unsigned short* Wb, int n) {
    int id = blockIdx.x * blockDim.x + threadIdx.x;
    if (id < n) Wb[id] = f2bf(W[id]);
}

// ---------------- channelwise LayerNorm + PReLU -> y (bf16, [B,C,T]) ----------
__global__ void k_ln(const float* __restrict__ x, const float* __restrict__ gamma,
                     const float* __restrict__ beta, const float* __restrict__ prelu,
                     unsigned short* __restrict__ y_bf) {
    int tid = threadIdx.x;
    int b  = blockIdx.x / (TT / 256);
    int t0 = (blockIdx.x % (TT / 256)) * 256;
    size_t base = (size_t)b * CC * TT + t0 + tid;
    float sum = 0.f, sq = 0.f;
    for (int c = 0; c < CC; ++c) {
        float v = x[base + (size_t)c * TT];
        sum += v; sq = fmaf(v, v, sq);
    }
    float mean = sum * (1.0f / CC);
    float var  = sq * (1.0f / CC) - mean * mean;
    float rstd = rsqrtf(var + 1e-8f);
    for (int c = 0; c < CC; ++c) {
        float v  = x[base + (size_t)c * TT];          // L0/L2 hot re-read
        float yv = (v - mean) * rstd * gamma[c] + beta[c];
        yv = yv >= 0.f ? yv : prelu[c] * yv;
        y_bf[base + (size_t)c * TT] = f2bf(yv);
    }
}

// ---------------- scan phase 1: chunk-local end states ------------------------
__global__ void k_scan1(const unsigned short* __restrict__ y_bf,
                        const float4* __restrict__ pw, float2* __restrict__ F) {
    int wid  = blockIdx.x * (blockDim.x >> 5) + (threadIdx.x >> 5);
    int lane = threadIdx.x & 31;
    int chunk = wid % NCHUNK;
    int bc    = wid / NCHUNK;          // b*CC + c
    int c     = bc % CC;
    float4 p = pw[c * NH + lane];
    float wr = p.x, wi = p.y;
    float sr = 0.f, si = 0.f;
    const unsigned short* yp = y_bf + (size_t)bc * TT + chunk * LCHUNK;
    for (int g = 0; g < LCHUNK; g += 32) {
        float yl = bf2f(yp[g + lane]);
        #pragma unroll
        for (int j = 0; j < 32; ++j) {
            float yj  = __shfl(yl, j, 32);
            float nsr = fmaf(wr, sr, fmaf(-wi, si, yj));
            float nsi = fmaf(wr, si, wi * sr);
            sr = nsr; si = nsi;
        }
    }
    F[((size_t)bc * NCHUNK + chunk) * NH + lane] = make_float2(sr, si);
}

// ---------------- scan phase 2: prefix states across chunks -------------------
__global__ void k_scan2(const float2* __restrict__ F, const float2* __restrict__ pwL,
                        float2* __restrict__ P) {
    int id = blockIdx.x * blockDim.x + threadIdx.x;   // bc*32 + n
    if (id >= BB * CC * NH) return;
    int n  = id & 31;
    int bc = id >> 5;
    int c  = bc % CC;
    float2 wl = pwL[c * NH + n];
    float pr = 0.f, pi = 0.f;
    for (int k = 0; k < NCHUNK; ++k) {
        P[((size_t)bc * NCHUNK + k) * NH + n] = make_float2(pr, pi);
        float2 f = F[((size_t)bc * NCHUNK + k) * NH + n];
        float npr = wl.x * pr - wl.y * pi + f.x;
        float npi = wl.x * pi + wl.y * pr + f.y;
        pr = npr; pi = npi;
    }
}

// ---------------- scan phase 3: outputs + D-skip + GELU -> u (bf16, [C, B*T]) -
// Mode-sum reduction via in-wave LDS transpose: lane n writes its row of 32
// per-step contributions (8x ds_store_b128), then reads column `lane`
// (conflict-free: row stride padded to 36 floats) and sums it serially.
#define XROW 36
__global__ void __launch_bounds__(256)
k_scan3(const unsigned short* __restrict__ y_bf,
        const float4* __restrict__ pw, const float2* __restrict__ P,
        const float* __restrict__ Dvec, unsigned short* __restrict__ u_bf) {
    __shared__ float xpose[8 * 32 * XROW];            // 36 KB, per-wave 4.5 KB slice
    int wid  = blockIdx.x * (blockDim.x >> 5) + (threadIdx.x >> 5);
    int lane = threadIdx.x & 31;
    int wv   = threadIdx.x >> 5;
    float* wlds = xpose + wv * 32 * XROW;
    int chunk = wid % NCHUNK;
    int bc    = wid / NCHUNK;
    int c     = bc % CC;
    int b     = bc / CC;
    float4 p = pw[c * NH + lane];
    float wr = p.x, wi = p.y, ccr = p.z, cci = p.w;
    float2 st = P[((size_t)bc * NCHUNK + chunk) * NH + lane];
    float sr = st.x, si = st.y;
    float dv = Dvec[c];
    const unsigned short* yp = y_bf + (size_t)bc * TT + chunk * LCHUNK;
    unsigned short* up = u_bf + (size_t)c * NTOT + (size_t)b * TT + chunk * LCHUNK;
    for (int g = 0; g < LCHUNK; g += 32) {
        float yl = bf2f(yp[g + lane]);
        float creg[32];
        #pragma unroll
        for (int j = 0; j < 32; ++j) {
            float yj  = __shfl(yl, j, 32);
            float nsr = fmaf(wr, sr, fmaf(-wi, si, yj));
            float nsi = fmaf(wr, si, wi * sr);
            sr = nsr; si = nsi;
            creg[j] = fmaf(ccr, sr, -cci * si);       // mode contribution at step j
        }
        // row write: lane n -> M[n][0..31]
        float4* rowp = (float4*)(wlds + lane * XROW);
        #pragma unroll
        for (int q = 0; q < 8; ++q)
            rowp[q] = make_float4(creg[4*q], creg[4*q+1], creg[4*q+2], creg[4*q+3]);
        __builtin_amdgcn_wave_barrier();              // same-wave LDS is in-order; block compiler reorder
        // column read: out[j=lane] = sum_n M[n][lane]
        float acc = 0.f;
        #pragma unroll
        for (int n = 0; n < 32; ++n) acc += wlds[n * XROW + lane];
        __builtin_amdgcn_wave_barrier();
        float sval = fmaf(dv, yl, acc);
        float gel  = 0.5f * sval * (1.0f + erff(sval * 0.70710678118f));
        up[g + lane] = f2bf(gel);                     // contiguous bf16 store
    }
}

// ---------------- WMMA GEMM [512x256]x[256x32] + GLU + residual ---------------
__global__ void __launch_bounds__(256)
k_gemm(const unsigned short* __restrict__ Wb, const unsigned short* __restrict__ u_bf,
       const float* __restrict__ b_out, const float* __restrict__ x,
       float* __restrict__ out) {
    __shared__ unsigned short ldsu[32 * 256];        // u tile TRANSPOSED [n][k], 16 KB
    const int tid  = threadIdx.x;
    const int lane = tid & 31;
    const int wv   = tid >> 5;                       // 8 waves
    const int nb   = blockIdx.x * 32;                // column block (32 t's, single b)
    const int b    = nb / TT;
    const int t0   = nb - b * TT;
    // stage: thread k reads row k (64B coalesced), scatters into column k of ldsu
    {
        union { uint4 q[4]; unsigned short s[32]; } row;
        const uint4* src = (const uint4*)(u_bf + (size_t)tid * NTOT + nb);
        row.q[0] = src[0]; row.q[1] = src[1]; row.q[2] = src[2]; row.q[3] = src[3];
        #pragma unroll
        for (int n = 0; n < 32; ++n) ldsu[n * 256 + tid] = row.s[n];
    }
    __syncthreads();
    const int hi  = (lane >> 4) & 1;
    const int nlo = lane & 15;
    v8f acc[2][2][2];                                // [row-tile][glu-half][col-tile]
    #pragma unroll
    for (int rp = 0; rp < 2; ++rp) {
        int r0 = (wv * 2 + rp) * 16;
        #pragma unroll
        for (int hf = 0; hf < 2; ++hf)
            #pragma unroll
            for (int ct = 0; ct < 2; ++ct)
                #pragma unroll
                for (int r = 0; r < 8; ++r)
                    acc[rp][hf][ct][r] = b_out[r0 + hf * 256 + r + hi * 8];
    }
    for (int ks = 0; ks < 8; ++ks) {                 // K = 256 in steps of 32
        v16bf bf[2];
        #pragma unroll
        for (int ct = 0; ct < 2; ++ct) {             // B frag: elems 0-7 -> K=8*hi.., 8-15 -> K=16+8*hi.., N=nlo
            union { uint4 q[2]; v16bf v; } ub;       // 2x ds_load_b128, no packing
            const uint4* bp = (const uint4*)(ldsu + (ct * 16 + nlo) * 256 + ks * 32 + hi * 8);
            ub.q[0] = bp[0]; ub.q[1] = bp[2];
            bf[ct] = ub.v;
        }
        #pragma unroll
        for (int rp = 0; rp < 2; ++rp) {
            int r0 = (wv * 2 + rp) * 16;
            #pragma unroll
            for (int hf = 0; hf < 2; ++hf) {         // A frag: elems 0-7 -> K=ko.., 8-15 -> K=ko+16..
                union { uint4 q[2]; v16bf v; } ua;
                const unsigned short* wp =
                    Wb + (size_t)(r0 + hf * 256 + nlo) * 256 + ks * 32 + hi * 8;
                ua.q[0] = *(const uint4*)wp;
                ua.q[1] = *(const uint4*)(wp + 16);
                #pragma unroll
                for (int ct = 0; ct < 2; ++ct)
                    acc[rp][hf][ct] = __builtin_amdgcn_wmma_f32_16x16x32_bf16(
                        false, ua.v, false, bf[ct], (short)0, acc[rp][hf][ct],
                        false, false);
            }
        }
    }
    // epilogue: GLU(z1, z2) + residual (C/D layout: elem r -> M=r+8*hi, N=nlo)
    #pragma unroll
    for (int rp = 0; rp < 2; ++rp) {
        int r0 = (wv * 2 + rp) * 16;
        #pragma unroll
        for (int ct = 0; ct < 2; ++ct) {
            int t = t0 + ct * 16 + nlo;
            #pragma unroll
            for (int r = 0; r < 8; ++r) {
                int crow = r0 + r + hi * 8;
                float z1 = acc[rp][0][ct][r];
                float z2 = acc[rp][1][ct][r];
                float sig = 1.0f / (1.0f + expf(-z2));
                size_t adr = ((size_t)b * CC + crow) * TT + t;
                out[adr] = x[adr] + z1 * sig;
            }
        }
    }
}

extern "C" void kernel_launch(void* const* d_in, const int* in_sizes, int n_in,
                              void* d_out, int out_size, void* d_ws, size_t ws_size,
                              hipStream_t stream) {
    const float* x      = (const float*)d_in[0];
    const float* gamma  = (const float*)d_in[1];
    const float* beta   = (const float*)d_in[2];
    const float* prelu  = (const float*)d_in[3];
    const float* log_dt = (const float*)d_in[4];
    const float* logAr  = (const float*)d_in[5];
    const float* Aim    = (const float*)d_in[6];
    const float* Cre    = (const float*)d_in[7];
    const float* Cim    = (const float*)d_in[8];
    const float* Dv     = (const float*)d_in[9];
    const float* Wout   = (const float*)d_in[10];
    const float* bout   = (const float*)d_in[11];
    float* out = (float*)d_out;

    char* ws = (char*)d_ws;
    size_t o = 0;
    unsigned short* y_bf = (unsigned short*)(ws + o); o += (size_t)BB * CC * TT * 2;
    unsigned short* u_bf = (unsigned short*)(ws + o); o += (size_t)CC * NTOT * 2;
    unsigned short* Wb   = (unsigned short*)(ws + o); o += (size_t)512 * 256 * 2;
    float4* pw  = (float4*)(ws + o); o += (size_t)CC * NH * sizeof(float4);
    float2* pwL = (float2*)(ws + o); o += (size_t)CC * NH * sizeof(float2);
    float2* F   = (float2*)(ws + o); o += (size_t)BB * CC * NCHUNK * NH * sizeof(float2);
    float2* P   = (float2*)(ws + o); o += (size_t)BB * CC * NCHUNK * NH * sizeof(float2);

    hipLaunchKernelGGL(k_params, dim3((CC * NH + 255) / 256), dim3(256), 0, stream,
                       log_dt, logAr, Aim, Cre, Cim, pw, pwL);
    hipLaunchKernelGGL(k_wcvt, dim3((512 * 256 + 255) / 256), dim3(256), 0, stream,
                       Wout, Wb, 512 * 256);
    hipLaunchKernelGGL(k_ln, dim3(BB * TT / 256), dim3(256), 0, stream,
                       x, gamma, beta, prelu, y_bf);
    int nwaves = BB * CC * NCHUNK;                   // 32768 waves
    hipLaunchKernelGGL(k_scan1, dim3(nwaves / 8), dim3(256), 0, stream, y_bf, pw, F);
    hipLaunchKernelGGL(k_scan2, dim3((BB * CC * NH + 255) / 256), dim3(256), 0, stream,
                       F, pwL, P);
    hipLaunchKernelGGL(k_scan3, dim3(nwaves / 8), dim3(256), 0, stream,
                       y_bf, pw, P, Dv, u_bf);
    hipLaunchKernelGGL(k_gemm, dim3(NTOT / 32), dim3(256), 0, stream,
                       Wb, u_bf, bout, x, out);
}
